// MultiSimilarityLossOriginal_12395275616922
// MI455X (gfx1250) — matrix operand order repaired and checked by reading, not compiled, not run on hardware
//
#include <hip/hip_runtime.h>
#include <hip/hip_bf16.h>
#include <math.h>

// ---------------------------------------------------------------------------
// Multi-Similarity loss, fused:  sim = F F^T  (16384x16384) never materialized.
// sim tiles via v_wmma_f32_16x16x32_f16 with f16 hi/lo split (3 WMMAs / K-step)
// for ~2^-21 dot-product accuracy (needed: neg term is exp(40*(sim-0.5))).
// ---------------------------------------------------------------------------

#define BATCH 16384
#define DIM   128
#define COLCHUNKS 8
#define TILES_PER_CHUNK ((BATCH / 16) / COLCHUNKS)   // 128 column tiles / chunk

typedef __attribute__((ext_vector_type(16))) _Float16 v16h;
typedef __attribute__((ext_vector_type(8)))  _Float16 v8h;
typedef __attribute__((ext_vector_type(8)))  float    v8f;

// exp(scale*(sim-thresh)) = exp2(sim*K1 + K0), constants fold log2(e):
#define LOG2E 1.4426950408889634f
#define KP1 (-2.0f * LOG2E)           // pos: exp(-2*(sim-0.5))
#define KP0 ( 2.0f * 0.5f * LOG2E)
#define KN1 ( 40.0f * LOG2E)          // neg: exp(40*(sim-0.5))
#define KN0 (-40.0f * 0.5f * LOG2E)

static __device__ __forceinline__ v16h wmma_f16(v16h a, v16h b, v8f& c) {
  c = __builtin_amdgcn_wmma_f32_16x16x32_f16(
      /*neg_a=*/false, a, /*neg_b=*/false, b,
      /*c_mod=*/(short)0, c, /*reuse_a=*/false, /*reuse_b=*/false);
  return a;
}

// A-matrix per-lane gather (16-bit 16x32 layout, ISA 7.12.2):
//   lane: m = lane&15, khalf = lane>>4
//   half h: K = 32*t + 16*(h>>3) + 8*khalf + (h&7)
// => two contiguous 8-half (16B) chunks per K-step.
static __device__ __forceinline__ v16h load_A(const _Float16* __restrict__ row,
                                              int t, int khalf) {
  const v8h c0 = *(const v8h*)(row + 32 * t + 8 * khalf);
  const v8h c1 = *(const v8h*)(row + 32 * t + 16 + 8 * khalf);
  v16h r;
#pragma unroll
  for (int i = 0; i < 8; ++i) { r[i] = c0[i]; r[i + 8] = c1[i]; }
  return r;
}

// --- Kernel 1: f32 -> (hi,lo) f16 split ------------------------------------
__global__ void __launch_bounds__(256)
prep_split(const float* __restrict__ feats, _Float16* __restrict__ fhi,
           _Float16* __restrict__ flo, int n) {
  int i = blockIdx.x * 256 + threadIdx.x;
  if (i < n) {
    float x = feats[i];
    _Float16 h = (_Float16)x;
    fhi[i] = h;
    flo[i] = (_Float16)(x - (float)h);
  }
}

// --- Kernel 2: fused tile GEMM + exp epilogue ------------------------------
// block = 256 threads = 8 waves. chunk = blockIdx&7 (shared by all 8 waves so
// B-tile streams hit WGP$); wave w handles row strip (blockIdx>>3)*8 + w.
__global__ void __launch_bounds__(256)
ms_tiles(const _Float16* __restrict__ fhi, const _Float16* __restrict__ flo,
         const int* __restrict__ labels, float* __restrict__ pos_sums,
         float* __restrict__ neg_sums) {
  const int lane  = threadIdx.x & 31;
  const int wave  = threadIdx.x >> 5;
  const int chunk = blockIdx.x & (COLCHUNKS - 1);
  const int strip = (blockIdx.x >> 3) * 8 + wave;
  const int rowBase = strip * 16;

  const int m     = lane & 15;
  const int khalf = lane >> 4;

  // Preload this wave's 16x128 A strip (hi+lo) into registers: 4 K-steps.
  const _Float16* arow_hi = fhi + (size_t)(rowBase + m) * DIM;
  const _Float16* arow_lo = flo + (size_t)(rowBase + m) * DIM;
  v16h Ahi[4], Alo[4];
#pragma unroll
  for (int t = 0; t < 4; ++t) {
    Ahi[t] = load_A(arow_hi, t, khalf);
    Alo[t] = load_A(arow_lo, t, khalf);
  }

  // Row labels / indices for the 8 C rows this lane owns (M = r + 8*khalf).
  int rl[8], rowIdx[8];
#pragma unroll
  for (int r = 0; r < 8; ++r) {
    rowIdx[r] = rowBase + r + 8 * khalf;
    rl[r] = labels[rowIdx[r]];
  }

  float pos[8], neg[8];
#pragma unroll
  for (int r = 0; r < 8; ++r) { pos[r] = 0.0f; neg[r] = 0.0f; }

  const int n  = lane & 15;   // C column within tile == B column
  const int kb = lane >> 4;   // B K-half select

  for (int ct = 0; ct < TILES_PER_CHUNK; ++ct) {
    const int jbase = chunk * (TILES_PER_CHUNK * 16) + ct * 16;
    const int col = jbase + n;
    // B per lane: column `col`, K = 32*t + 16*(h>>3) + 8*kb + (h&7) (same as A).
    const _Float16* bcol_hi = fhi + (size_t)col * DIM;
    const _Float16* bcol_lo = flo + (size_t)col * DIM;
    v16h Bhi[4], Blo[4];
#pragma unroll
    for (int t = 0; t < 4; ++t) {
      Bhi[t] = load_A(bcol_hi, t, kb);
      Blo[t] = load_A(bcol_lo, t, kb);
    }
    const int lc = labels[col];

    v8f c = {};
#pragma unroll
    for (int t = 0; t < 4; ++t) {
      wmma_f16(Ahi[t], Bhi[t], c);   // hi*hi
      wmma_f16(Ahi[t], Blo[t], c);   // hi*lo
      wmma_f16(Alo[t], Bhi[t], c);   // lo*hi  (lo*lo ~2^-22, dropped)
    }

    // Epilogue: route each sim through pos/neg exp and accumulate.
#pragma unroll
    for (int r = 0; r < 8; ++r) {
      const float sim  = c[r];
      const bool  same = (rl[r] == lc);
      const float k1 = same ? KP1 : KN1;
      const float k0 = same ? KP0 : KN0;
      float e = exp2f(fmaf(sim, k1, k0));
      const bool self = same && (col == rowIdx[r]);
      e = self ? 0.0f : e;
      pos[r] += same ? e : 0.0f;
      neg[r] += same ? 0.0f : e;
    }
  }

  // Reduce across the 16 lanes sharing each C row (xor masks stay in-half),
  // then one atomicAdd per row (lane 0 -> rows M=0..7, lane 16 -> M=8..15).
#pragma unroll
  for (int r = 0; r < 8; ++r) {
    float p = pos[r], q = neg[r];
#pragma unroll
    for (int mk = 1; mk < 16; mk <<= 1) {
      p += __shfl_xor(p, mk, 32);
      q += __shfl_xor(q, mk, 32);
    }
    if ((lane & 15) == 0) {
      atomicAdd(&pos_sums[rowIdx[r]], p);
      atomicAdd(&neg_sums[rowIdx[r]], q);
    }
  }
}

// --- Kernel 3: per-row log1p + scalar reduction ----------------------------
// exp(.) > 0 always, so pos_sum>0 <=> row has a positive pair (valid check).
__global__ void __launch_bounds__(256)
finalize(const float* __restrict__ pos_sums, const float* __restrict__ neg_sums,
         float* __restrict__ out) {
  __shared__ float sdata[256];
  float acc = 0.0f;
  for (int i = threadIdx.x; i < BATCH; i += 256) {
    const float p = pos_sums[i];
    const float q = neg_sums[i];
    if (p > 0.0f && q > 0.0f)
      acc += 0.5f * log1pf(p) + 0.025f * log1pf(q);  // /SCALE_POS, /SCALE_NEG
  }
  sdata[threadIdx.x] = acc;
  __syncthreads();
  for (int s = 128; s > 0; s >>= 1) {
    if (threadIdx.x < (unsigned)s) sdata[threadIdx.x] += sdata[threadIdx.x + s];
    __syncthreads();
  }
  if (threadIdx.x == 0) out[0] = sdata[0] / (float)BATCH;
}

// ---------------------------------------------------------------------------
extern "C" void kernel_launch(void* const* d_in, const int* in_sizes, int n_in,
                              void* d_out, int out_size, void* d_ws,
                              size_t ws_size, hipStream_t stream) {
  (void)in_sizes; (void)n_in; (void)out_size; (void)ws_size;
  const int*   labels = (const int*)d_in[0];
  const float* feats  = (const float*)d_in[1];
  float* out = (float*)d_out;

  char* ws = (char*)d_ws;
  float*    pos_sums = (float*)ws;                       //  64 KB
  float*    neg_sums = (float*)(ws + 64 * 1024);         //  64 KB
  _Float16* fhi = (_Float16*)(ws + 128 * 1024);          //   4 MB
  _Float16* flo = (_Float16*)(ws + 128 * 1024 +
                              (size_t)BATCH * DIM * sizeof(_Float16)); // 4 MB

  hipMemsetAsync(pos_sums, 0, 2 * (size_t)BATCH * sizeof(float), stream);

  const int nelem = BATCH * DIM;
  prep_split<<<(nelem + 255) / 256, 256, 0, stream>>>(feats, fhi, flo, nelem);

  // 1024 row strips x 8 column chunks = 8192 waves -> 1024 blocks of 8 waves.
  ms_tiles<<<(BATCH / 16 / 8) * COLCHUNKS, 256, 0, stream>>>(
      fhi, flo, labels, pos_sums, neg_sums);

  finalize<<<1, 256, 0, stream>>>(pos_sums, neg_sums, out);
}
